// MultiHeadSelfAttention_5634997093111
// MI455X (gfx1250) — matrix operand order repaired and checked by reading, not compiled, not run on hardware
//
#include <hip/hip_runtime.h>

#define D_MODEL 768
#define NHEAD   12
#define HD      64
#define SEQ     2048
#define BATCH   2
#define MROWS   (BATCH*SEQ)   // 4096

typedef __attribute__((ext_vector_type(16))) _Float16 v16h;
typedef __attribute__((ext_vector_type(8)))  _Float16 v8h;
typedef __attribute__((ext_vector_type(8)))  float    v8f;

__device__ __forceinline__ v8f wmma_f16(v16h a, v16h b, v8f c) {
  // D = A(16x32 f16) * B(32x16 f16) + C(16x16 f32)
  return __builtin_amdgcn_wmma_f32_16x16x32_f16(false, a, false, b,
                                                (short)0, c, false, false);
}

// Combine two contiguous 16B half-vectors into one fragment register set.
__device__ __forceinline__ v16h make_frag(v8h lo, v8h hi) {
  v16h f;
#pragma unroll
  for (int i = 0; i < 8; ++i) { f[i] = lo[i]; f[i + 8] = hi[i]; }
  return f;
}

// A-fragment (16x32, f16), ISA layout:
// lane L<16: M=L, K={0..7,16..23}; lane L>=16: M=L-16, K={8..15,24..31}
__device__ __forceinline__ v16h load_frag_a(const _Float16* base, int ld) {
  const int lane = threadIdx.x & 31;
  const int M    = lane & 15;
  const int kb   = (lane >> 4) << 3;       // 0 or 8
  const _Float16* p = base + (size_t)M * ld + kb;
  return make_frag(*(const v8h*)p, *(const v8h*)(p + 16));
}

// B-fragment (32x16, f16): element (K,N) at base[N*ld + K];
// lane: N = L&15, K = {kb..kb+7, kb+16..kb+23}, kb = 8*(L>=16)  -> two 16B loads.
__device__ __forceinline__ v16h load_frag_b(const _Float16* base, int ld) {
  const int lane = threadIdx.x & 31;
  const int N    = lane & 15;
  const int kb   = (lane >> 4) << 3;       // 0 or 8
  const _Float16* p = base + (size_t)N * ld + kb;
  return make_frag(*(const v8h*)p, *(const v8h*)(p + 16));
}

__global__ void cast_f32_to_f16(const float* __restrict__ src,
                                _Float16* __restrict__ dst, int n) {
  int i = blockIdx.x * blockDim.x + threadIdx.x;
  if (i < n) dst[i] = (_Float16)src[i];
}

// Y = A(4096x768) @ W(768x768)^T. One wave -> 32 rows x 64 cols.
// Ping-pong double-buffered K loop (no register-rotation copies).
// MODE 0: f16 [B,H,S,hd] * 1/8 (Q)   MODE 1: f16 [B,H,S,hd] (K)
// MODE 2: f16 [B,H,hd,S]  (V^T)      MODE 3: f32 row-major (final proj)
template <int MODE>
__global__ void __launch_bounds__(128, 1)
gemm_wmma(const _Float16* __restrict__ A,
          const _Float16* __restrict__ W,
          _Float16* __restrict__ outH,
          float* __restrict__ outF) {
  constexpr int K = D_MODEL;               // 768
  constexpr int N = D_MODEL;               // 768

  const int wid = (blockIdx.x * blockDim.x + threadIdx.x) >> 5;
  const int wm  = wid / (N >> 6);
  const int wn  = wid % (N >> 6);
  const int m0  = wm << 5;                 // 32 rows per wave
  const int n0  = wn << 6;                 // 64 cols per wave

  // ---- per-lane fragment base pointers (k indexed via immediate offsets) ----
  const int lane = threadIdx.x & 31;
  const int lo16 = lane & 15;
  const int kbA  = (lane >> 4) << 3;       // 0 or 8
  const int kbB  = (lane >> 4) << 3;       // 0 or 8
  const _Float16* pA0 = A + (size_t)(m0 + lo16) * K + kbA;
  const _Float16* pA1 = pA0 + (size_t)16 * K;
  const _Float16* pB0 = W + (size_t)(n0 + lo16) * K + kbB;

  v8f acc0[4], acc1[4];
#pragma unroll
  for (int c = 0; c < 4; ++c)
#pragma unroll
    for (int i = 0; i < 8; ++i) { acc0[c][i] = 0.0f; acc1[c][i] = 0.0f; }

  auto ldA = [&](const _Float16* p, int k) {
    return make_frag(*(const v8h*)(p + k), *(const v8h*)(p + k + 16));
  };
  auto ldB = [&](int c, int k) {
    const _Float16* p = pB0 + (size_t)(c * 16) * K + k;
    return make_frag(*(const v8h*)p, *(const v8h*)(p + 16));
  };
  auto mm = [&](const v16h& x0, const v16h& x1, const v16h (&bb)[4]) {
#pragma unroll
    for (int c = 0; c < 4; ++c) {
      acc0[c] = wmma_f16(x0, bb[c], acc0[c]);
      acc1[c] = wmma_f16(x1, bb[c], acc1[c]);
    }
  };

  // ---- prologue: even buffers at k = 0 ----
  v16h aE0 = ldA(pA0, 0), aE1 = ldA(pA1, 0);
  v16h bE[4];
#pragma unroll
  for (int c = 0; c < 4; ++c) bE[c] = ldB(c, 0);

  v16h aO0, aO1, bO[4];
  int k0 = 0;
  // ---- ping-pong main loop: 2 K-steps per iteration, no buffer copies ----
  for (; k0 < K - 64; k0 += 64) {
    __builtin_prefetch(pA0 + k0 + 256, 0, 1);        // global_prefetch_b8
    __builtin_prefetch(pB0 + k0 + 256, 0, 1);
    aO0 = ldA(pA0, k0 + 32); aO1 = ldA(pA1, k0 + 32);
#pragma unroll
    for (int c = 0; c < 4; ++c) bO[c] = ldB(c, k0 + 32);
    mm(aE0, aE1, bE);                                 // compute k0 (even)
    aE0 = ldA(pA0, k0 + 64); aE1 = ldA(pA1, k0 + 64);
#pragma unroll
    for (int c = 0; c < 4; ++c) bE[c] = ldB(c, k0 + 64);
    mm(aO0, aO1, bO);                                 // compute k0+32 (odd)
  }
  // ---- tail: k0 == K-64; last odd load + two computes ----
  aO0 = ldA(pA0, k0 + 32); aO1 = ldA(pA1, k0 + 32);
#pragma unroll
  for (int c = 0; c < 4; ++c) bO[c] = ldB(c, k0 + 32);
  mm(aE0, aE1, bE);
  mm(aO0, aO1, bO);

  // ---- store 2 row-tiles x 4 col-tiles ----
  const int Nc   = lo16;
  const int roff = (lane >> 4) << 3;       // 0 or 8
#pragma unroll
  for (int half = 0; half < 2; ++half) {
    const int mbase = m0 + half * 16;
#pragma unroll
    for (int c = 0; c < 4; ++c) {
      const int n = n0 + c * 16 + Nc;
#pragma unroll
      for (int r = 0; r < 8; ++r) {
        const int m = mbase + roff + r;
        const float v = half ? acc1[c][r] : acc0[c][r];
        if constexpr (MODE == 3) {
          outF[(size_t)m * N + n] = v;
        } else {
          const int b = m >> 11, s = m & 2047;   // S = 2048
          const int h = n >> 6,  d = n & 63;     // hd = 64
          if constexpr (MODE == 0)
            outH[((size_t)(b * NHEAD + h) * SEQ + s) * HD + d] = (_Float16)(v * 0.125f);
          else if constexpr (MODE == 1)
            outH[((size_t)(b * NHEAD + h) * SEQ + s) * HD + d] = (_Float16)v;
          else
            outH[((size_t)(b * NHEAD + h) * HD + d) * SEQ + s] = (_Float16)v;
        }
      }
    }
  }
}

// Flash-style causal attention. One wave per 16-query tile per (b,h).
// Q: [B,H,S,hd] (pre-scaled), Kh: [B,H,S,hd], VT: [B,H,hd,S], O: [B,S,D] f16.
__global__ void __launch_bounds__(128, 1)
attn_kernel(const _Float16* __restrict__ Q,
            const _Float16* __restrict__ Kh,
            const _Float16* __restrict__ VT,
            _Float16* __restrict__ O) {
  __shared__ _Float16 pbuf[4][16 * 32];    // per-wave P staging (1 KB each)

  const int lane = threadIdx.x & 31;
  const int wib  = threadIdx.x >> 5;
  const int tix  = blockIdx.x * 4 + wib;   // 3072 waves total
  const int qt   = tix & 127;              // S/16 = 128 q-tiles
  const int bh   = tix >> 7;               // B*H = 24
  const int q0   = qt << 4;

  const _Float16* qb = Q  + (size_t)bh * SEQ * HD;
  const _Float16* kp = Kh + (size_t)bh * SEQ * HD;
  const _Float16* vb = VT + (size_t)bh * HD * SEQ;
  _Float16* myp = &pbuf[wib][0];

  const int hi  = lane >> 4;               // 0/1: which 8-row half
  const int col = lane & 15;

  const v16h qf0 = load_frag_a(qb + (size_t)q0 * HD, HD);
  const v16h qf1 = load_frag_a(qb + (size_t)q0 * HD + 32, HD);

  float m_run[8], l_run[8];
  v8f acc[4];
#pragma unroll
  for (int r = 0; r < 8; ++r) { m_run[r] = -1e30f; l_run[r] = 0.0f; }
#pragma unroll
  for (int c = 0; c < 4; ++c)
#pragma unroll
    for (int i = 0; i < 8; ++i) acc[c][i] = 0.0f;

  for (int kt0 = 0; kt0 <= q0 + 15; kt0 += 32) {
    // prefetch next chunk's K rows and V columns into cache
    __builtin_prefetch(kp + (size_t)(kt0 + 32) * HD, 0, 1);
    __builtin_prefetch(vb + kt0 + 32, 0, 1);

    // ---- scores for 32 keys: two 16x16 f32 tiles, K-dim = hd = 64 ----
    v8f sc[2];
#pragma unroll
    for (int t = 0; t < 2; ++t) {
      v8f z;
#pragma unroll
      for (int i = 0; i < 8; ++i) z[i] = 0.0f;
      v16h b0 = load_frag_b(kp + (size_t)(kt0 + t * 16) * HD, HD);
      z = wmma_f16(qf0, b0, z);
      v16h b1 = load_frag_b(kp + (size_t)(kt0 + t * 16) * HD + 32, HD);
      z = wmma_f16(qf1, b1, z);
      sc[t] = z;
    }

    // ---- causal mask + online softmax stats (row reductions in 16-lane halves) ----
    float mnew[8];
#pragma unroll
    for (int r = 0; r < 8; ++r) {
      const int qrow = q0 + r + hi * 8;
      float s0 = (kt0 + col      <= qrow) ? sc[0][r] : -1e30f;
      float s1 = (kt0 + 16 + col <= qrow) ? sc[1][r] : -1e30f;
      sc[0][r] = s0; sc[1][r] = s1;
      float mx = fmaxf(s0, s1);
      mx = fmaxf(mx, __shfl_xor(mx, 1, 32));
      mx = fmaxf(mx, __shfl_xor(mx, 2, 32));
      mx = fmaxf(mx, __shfl_xor(mx, 4, 32));
      mx = fmaxf(mx, __shfl_xor(mx, 8, 32));
      mnew[r] = fmaxf(m_run[r], mx);
    }

#pragma unroll
    for (int r = 0; r < 8; ++r) {
      const float e0 = __expf(sc[0][r] - mnew[r]);
      const float e1 = __expf(sc[1][r] - mnew[r]);
      const int qr = r + hi * 8;
      myp[qr * 32 + col]      = (_Float16)e0;
      myp[qr * 32 + 16 + col] = (_Float16)e1;
      float rs = e0 + e1;
      rs += __shfl_xor(rs, 1, 32);
      rs += __shfl_xor(rs, 2, 32);
      rs += __shfl_xor(rs, 4, 32);
      rs += __shfl_xor(rs, 8, 32);
      const float scale = __expf(m_run[r] - mnew[r]);
      l_run[r] = l_run[r] * scale + rs;
      m_run[r] = mnew[r];
#pragma unroll
      for (int c = 0; c < 4; ++c) acc[c][r] *= scale;
    }

    // LDS writes by other lanes must land before cross-lane A-fragment reads.
    // Same-wave LDS ops are issued in order but complete async -> wait DScnt.
    asm volatile("s_wait_dscnt 0x0" ::: "memory");

    const v16h pf = load_frag_a(myp, 32);  // P as A-fragment (16x32, keys = K-dim)

    // ---- acc += P @ V, 4 chunks of 16 output features ----
#pragma unroll
    for (int c = 0; c < 4; ++c) {
      v16h vf = load_frag_b(vb + (size_t)(c * 16) * SEQ + kt0, SEQ);
      acc[c] = wmma_f16(pf, vf, acc[c]);
    }
  }

  // ---- finalize: divide by row sums, store f16 [B,S,D] ----
  float inv[8];
#pragma unroll
  for (int r = 0; r < 8; ++r) inv[r] = 1.0f / l_run[r];
  const int b = bh / NHEAD, h = bh % NHEAD;
  _Float16* ob = O + (size_t)b * SEQ * D_MODEL + h * HD;
#pragma unroll
  for (int c = 0; c < 4; ++c)
#pragma unroll
    for (int r = 0; r < 8; ++r) {
      const int qrow = q0 + r + hi * 8;
      ob[(size_t)qrow * D_MODEL + c * 16 + col] = (_Float16)(acc[c][r] * inv[r]);
    }
}

extern "C" void kernel_launch(void* const* d_in, const int* in_sizes, int n_in,
                              void* d_out, int out_size, void* d_ws, size_t ws_size,
                              hipStream_t stream) {
  const float* x  = (const float*)d_in[0];
  const float* Wq = (const float*)d_in[1];
  const float* Wk = (const float*)d_in[2];
  const float* Wv = (const float*)d_in[3];
  const float* Wo = (const float*)d_in[4];

  const size_t szX = (size_t)MROWS * D_MODEL;      // 3,145,728 elems
  const size_t szW = (size_t)D_MODEL * D_MODEL;    //   589,824 elems

  _Float16* Xh  = (_Float16*)d_ws;
  _Float16* Wqh = Xh  + szX;
  _Float16* Wkh = Wqh + szW;
  _Float16* Wvh = Wkh + szW;
  _Float16* Woh = Wvh + szW;
  _Float16* Qh  = Woh + szW;                       // [B,H,S,hd], pre-scaled
  _Float16* Kh  = Qh  + szX;                       // [B,H,S,hd]
  _Float16* VT  = Kh  + szX;                       // [B,H,hd,S]
  _Float16* Oh  = VT  + szX;                       // [B,S,D]

  cast_f32_to_f16<<<(int)((szX + 255) / 256), 256, 0, stream>>>(x,  Xh,  (int)szX);
  cast_f32_to_f16<<<(int)((szW + 255) / 256), 256, 0, stream>>>(Wq, Wqh, (int)szW);
  cast_f32_to_f16<<<(int)((szW + 255) / 256), 256, 0, stream>>>(Wk, Wkh, (int)szW);
  cast_f32_to_f16<<<(int)((szW + 255) / 256), 256, 0, stream>>>(Wv, Wvh, (int)szW);
  cast_f32_to_f16<<<(int)((szW + 255) / 256), 256, 0, stream>>>(Wo, Woh, (int)szW);

  const int nwaves  = (MROWS / 32) * (D_MODEL / 64);  // 1536
  const int gblocks = nwaves / 4;                     // 128 threads = 4 waves/block

  gemm_wmma<0><<<gblocks, 128, 0, stream>>>(Xh, Wqh, Qh, nullptr);
  gemm_wmma<1><<<gblocks, 128, 0, stream>>>(Xh, Wkh, Kh, nullptr);
  gemm_wmma<2><<<gblocks, 128, 0, stream>>>(Xh, Wvh, VT, nullptr);

  // B*H*(S/16) = 3072 q-tile waves -> 768 blocks of 4 waves
  attn_kernel<<<768, 128, 0, stream>>>(Qh, Kh, VT, Oh);

  gemm_wmma<3><<<gblocks, 128, 0, stream>>>(Oh, Woh, nullptr, (float*)d_out);
}
